// MambaTransformer2_16604343567019
// MI455X (gfx1250) — matrix operand order repaired and checked by reading, not compiled, not run on hardware
//
#include <hip/hip_runtime.h>
#include <hip/hip_bf16.h>
#include <math.h>

// ---------------- model constants ----------------
#define Bc   8
#define Nc   768
#define Dc   512
#define Fc   3
#define Tc   256
#define DIc  1024
#define DSc  16
#define DTRc 32
#define Kc   4
#define DEPTHc 4
#define Hc   8
#define DHc  64
#define CBc  1024
#define BNc  (Bc * Nc)   // 6144 rows everywhere

typedef __bf16 bf16;
typedef __bf16 v16bf __attribute__((ext_vector_type(16)));
typedef __bf16 v8bf  __attribute__((ext_vector_type(8)));
typedef float  v8f   __attribute__((ext_vector_type(8)));

// epilogue modes
#define EPI_STORE         0
#define EPI_ADD           1
#define EPI_BIAS          2
#define EPI_BIAS_GELU     3
#define EPI_BIAS_SOFTPLUS 4
#define EPI_BIAS_ADD      5
#define EPI_OUTBIAS       6

__device__ __forceinline__ float gelu_exact(float v) {
    return 0.5f * v * (1.0f + erff(v * 0.70710678118654752f));
}
__device__ __forceinline__ float softplus_f(float v) {
    return v > 20.0f ? v : log1pf(__expf(v));
}
__device__ __forceinline__ float silu_f(float v) {
    return v / (1.0f + __expf(-v));
}

// CDNA5 async global->LDS copy (ASYNCcnt path, CDNA5 ISA)
__device__ __forceinline__ void async_load_b128(const void* gaddr, void* lds) {
    unsigned lds_off = (unsigned)(uintptr_t)lds;   // low 32 bits = LDS byte offset
    asm volatile("global_load_async_to_lds_b128 %0, %1, off"
                 :: "v"(lds_off), "v"(gaddr) : "memory");
}
__device__ __forceinline__ void wait_async() {
    asm volatile("s_wait_asynccnt 0x0" ::: "memory");
}

// ---------------------------------------------------------------------------
// BF16 WMMA GEMM: C[M,N] = A[M,K] * B[K,N] (+ fp32 epilogue), via
// V_WMMA_F32_16X16X32_BF16 (K=32 per issue, fp32 accumulate).
// A: compact bf16 (M x K) row-major.  Bt: compact bf16 (N x K) row-major
// (i.e. logical B pre-transposed), so every fragment load is a contiguous
// 16-byte chunk matching the ISA 7.12.2 16-bit layouts:
//   A lane (m = lane&15, h = lane>>4): elems 0-7 = K[k+8h .. k+8h+7],
//                                      elems 8-15 = K[k+16+8h .. k+16+8h+7]
//   B lane (n = lane&15, h = lane>>4): elems 0-7 = K[k+8h .. k+8h+7],
//                                      elems 8-15 = K[k+16+8h .. k+16+8h+7]
// One wave = one 16x16 tile; 8 waves / block.  M%16==0, N%16==0, K%32==0.
// ---------------------------------------------------------------------------
__global__ __launch_bounds__(256) void gemm_bf16_kernel(
    const bf16* __restrict__ A,
    const bf16* __restrict__ Bt,
    const float* __restrict__ bias, int mode, int biasld,
    const float* __restrict__ Cadd,
    float* __restrict__ C, int ldc,
    int M, int N, int K)
{
    const int lane = threadIdx.x & 31;
    const int wave = threadIdx.x >> 5;
    const int tilesN = N >> 4;
    const int tile = blockIdx.x * 8 + wave;
    if (tile >= (M >> 4) * tilesN) return;      // wave-uniform: EXEC stays full
    const int m0 = (tile / tilesN) << 4;
    const int n0 = (tile % tilesN) << 4;

    const int mi   = lane & 15;
    const int hi   = lane >> 4;
    const int aoff = hi << 3;    // 0 / 8
    const int boff = hi << 3;    // 0 / 8

    const bf16* arow = A  + (size_t)(m0 + mi) * K;
    const bf16* brow = Bt + (size_t)(n0 + mi) * K;

    v8f acc = {};
    for (int k = 0; k < K; k += 32) {
        if (k + 32 < K) {                       // WGP-scope stream-ahead prefetch
            __builtin_prefetch(arow + k + 32, 0, 3);
            __builtin_prefetch(brow + k + 32, 0, 3);
        }
        v8bf a0 = *(const v8bf*)(arow + k + aoff);
        v8bf a1 = *(const v8bf*)(arow + k + 16 + aoff);
        v8bf b0 = *(const v8bf*)(brow + k + boff);
        v8bf b1 = *(const v8bf*)(brow + k + boff + 16);
        v16bf av = __builtin_shufflevector(a0, a1, 0,1,2,3,4,5,6,7,8,9,10,11,12,13,14,15);
        v16bf bv = __builtin_shufflevector(b0, b1, 0,1,2,3,4,5,6,7,8,9,10,11,12,13,14,15);
        acc = __builtin_amdgcn_wmma_f32_16x16x32_bf16(
            false, av, false, bv, (short)0, acc, false, false);
    }

    const int col = n0 + mi;
    const int rbase = m0 + (hi << 3);
#pragma unroll
    for (int i = 0; i < 8; ++i) {
        const int row = rbase + i;
        float v = acc[i];
        switch (mode) {
            case EPI_ADD:           v += Cadd[(size_t)row * ldc + col]; break;
            case EPI_BIAS:          v += bias[col]; break;
            case EPI_BIAS_GELU:     v = gelu_exact(v + bias[col]); break;
            case EPI_BIAS_SOFTPLUS: v = softplus_f(v + bias[col]); break;
            case EPI_BIAS_ADD:      v += bias[col] + Cadd[(size_t)row * ldc + col]; break;
            case EPI_OUTBIAS:       v += bias[(size_t)(row % Nc) * biasld + col]; break;
            default: break;
        }
        C[(size_t)row * ldc + col] = v;
    }
}

// ---------------------------------------------------------------------------
// conversion kernels (f32 -> bf16)
// ---------------------------------------------------------------------------
// strided f32 (rows x ld, take first Kd cols) -> compact bf16 (rows x Kd)
__global__ void cvt_bf16_kernel(const float* __restrict__ in, int ld,
                                bf16* __restrict__ out, int rows, int Kd)
{
    size_t idx = (size_t)blockIdx.x * blockDim.x + threadIdx.x;
    if (idx >= (size_t)rows * Kd) return;
    int c = idx % Kd;
    size_t r = idx / Kd;
    out[idx] = (bf16)in[r * ld + c];
}

// batched transpose-convert: in (S, Kd, Nd) f32 -> out (S, Nd, Kd) bf16
__global__ void cvt_bf16_bt_kernel(const float* __restrict__ in,
                                   bf16* __restrict__ out, int S, int Kd, int Nd)
{
    size_t idx = (size_t)blockIdx.x * blockDim.x + threadIdx.x;
    if (idx >= (size_t)S * Kd * Nd) return;
    int k = idx % Kd;
    size_t r = idx / Kd;
    int n = r % Nd;
    int s = r / Nd;
    out[idx] = (bf16)in[((size_t)s * Kd + k) * Nd + n];
}

// ---------------------------------------------------------------------------
// elementwise / small kernels
// ---------------------------------------------------------------------------
__global__ void embed_kernel(const int* __restrict__ sM,
                             const float* __restrict__ tok,
                             const float* __restrict__ pos,
                             float* __restrict__ x)
{
    size_t idx = (size_t)blockIdx.x * blockDim.x + threadIdx.x;
    if (idx >= (size_t)BNc * Dc) return;
    int d = idx % Dc;
    size_t bn = idx / Dc;
    int n = bn % Nc;
    x[idx] = tok[(size_t)sM[bn] * Dc + d] + pos[(size_t)n * Dc + d];
}

__global__ __launch_bounds__(256) void rmsnorm_kernel(
    const float* __restrict__ x, const float* __restrict__ w,
    float* __restrict__ out, float eps)
{
    const int row = blockIdx.x;
    const float* xr = x + (size_t)row * Dc;
    float v0 = xr[threadIdx.x], v1 = xr[threadIdx.x + 256];
    __shared__ float red[256];
    red[threadIdx.x] = v0 * v0 + v1 * v1;
    __syncthreads();
    for (int s = 128; s > 0; s >>= 1) {
        if (threadIdx.x < s) red[threadIdx.x] += red[threadIdx.x + s];
        __syncthreads();
    }
    float sc = rsqrtf(red[0] / (float)Dc + eps);
    out[(size_t)row * Dc + threadIdx.x]       = v0 * sc * w[threadIdx.x];
    out[(size_t)row * Dc + threadIdx.x + 256] = v1 * sc * w[threadIdx.x + 256];
}

__global__ __launch_bounds__(256) void layernorm_kernel(
    const float* __restrict__ x, const float* __restrict__ w,
    const float* __restrict__ b, float* __restrict__ out, float eps)
{
    const int row = blockIdx.x;
    const float* xr = x + (size_t)row * Dc;
    float v0 = xr[threadIdx.x], v1 = xr[threadIdx.x + 256];
    __shared__ float rs[256], rq[256];
    rs[threadIdx.x] = v0 + v1;
    rq[threadIdx.x] = v0 * v0 + v1 * v1;
    __syncthreads();
    for (int s = 128; s > 0; s >>= 1) {
        if (threadIdx.x < s) { rs[threadIdx.x] += rs[threadIdx.x + s];
                               rq[threadIdx.x] += rq[threadIdx.x + s]; }
        __syncthreads();
    }
    float m = rs[0] / (float)Dc;
    float var = rq[0] / (float)Dc - m * m;
    float sc = rsqrtf(var + eps);
    out[(size_t)row * Dc + threadIdx.x] =
        (v0 - m) * sc * w[threadIdx.x] + b[threadIdx.x];
    out[(size_t)row * Dc + threadIdx.x + 256] =
        (v1 - m) * sc * w[threadIdx.x + 256] + b[threadIdx.x + 256];
}

// causal depthwise conv1d (K=4) over u = uz[:, :DI], then silu
__global__ void mamba_conv_kernel(const float* __restrict__ uz,
                                  const float* __restrict__ w,
                                  const float* __restrict__ cb,
                                  float* __restrict__ uc)
{
    size_t idx = (size_t)blockIdx.x * blockDim.x + threadIdx.x;
    if (idx >= (size_t)BNc * DIc) return;
    int c = idx % DIc;
    size_t bn = idx / DIc;
    int t = bn % Nc;
    int bb = bn / Nc;
    float s = cb[c];
#pragma unroll
    for (int j = 0; j < Kc; ++j) {
        int tt = t - (Kc - 1) + j;
        if (tt >= 0)
            s += w[c * Kc + j] * uz[((size_t)(bb * Nc + tt) * 2 * DIc) + c];
    }
    uc[idx] = silu_f(s);
}

// selective-scan: one thread per (batch, channel) holds DS=16 states
__global__ void mamba_scan_kernel(const float* __restrict__ dt,
                                  const float* __restrict__ uc,
                                  const float* __restrict__ xdb,
                                  const float* __restrict__ Alog,
                                  float* __restrict__ y)
{
    int gid = blockIdx.x * blockDim.x + threadIdx.x;
    if (gid >= Bc * DIc) return;
    int bb = gid / DIc, d = gid % DIc;
    float A[DSc], h[DSc];
#pragma unroll
    for (int s = 0; s < DSc; ++s) { A[s] = -__expf(Alog[d * DSc + s]); h[s] = 0.f; }
    for (int t = 0; t < Nc; ++t) {
        size_t bn = (size_t)bb * Nc + t;
        float dtv = dt[bn * DIc + d];
        float uv  = uc[bn * DIc + d];
        const float* Bv = xdb + bn * 64 + DTRc;
        const float* Cv = Bv + DSc;
        float du = dtv * uv;
        float acc = 0.f;
#pragma unroll
        for (int s = 0; s < DSc; ++s) {
            h[s] = __expf(dtv * A[s]) * h[s] + du * Bv[s];
            acc += h[s] * Cv[s];
        }
        y[bn * DIc + d] = acc;
    }
}

// y = (y + Dp[c]*u) * silu(z),  z = uz[:, DI:]
__global__ void gate_kernel(float* __restrict__ y,
                            const float* __restrict__ uc,
                            const float* __restrict__ uz,
                            const float* __restrict__ Dp)
{
    size_t idx = (size_t)blockIdx.x * blockDim.x + threadIdx.x;
    if (idx >= (size_t)BNc * DIc) return;
    int c = idx % DIc;
    size_t bn = idx / DIc;
    float z = uz[bn * 2 * DIc + DIc + c];
    y[idx] = (y[idx] + Dp[c] * uc[idx]) * silu_f(z);
}

__global__ void flip_kernel(const float* __restrict__ in, float* __restrict__ out, int L)
{
    size_t idx = (size_t)blockIdx.x * blockDim.x + threadIdx.x;
    if (idx >= (size_t)BNc * L) return;
    int d = idx % L;
    size_t bn = idx / L;
    int t = bn % Nc, bb = bn / Nc;
    out[((size_t)(bb * Nc + (Nc - 1 - t)) * L) + d] = in[idx];
}

__global__ void addflip_kernel(const float* __restrict__ in, float* __restrict__ x)
{
    size_t idx = (size_t)blockIdx.x * blockDim.x + threadIdx.x;
    if (idx >= (size_t)BNc * Dc) return;
    int d = idx % Dc;
    size_t bn = idx / Dc;
    int t = bn % Nc, bb = bn / Nc;
    x[idx] += in[((size_t)(bb * Nc + (Nc - 1 - t)) * Dc) + d];
}

__global__ void add_kernel(float* __restrict__ x, const float* __restrict__ s)
{
    size_t idx = (size_t)blockIdx.x * blockDim.x + threadIdx.x;
    if (idx >= (size_t)BNc * Dc) return;
    x[idx] += s[idx];
}

// streaming-softmax attention, one block per (batch-frame, head), one thread
// per query row; K/V staged through LDS via CDNA5 async global->LDS copies.
__global__ __launch_bounds__(256) void attn_kernel(const float* __restrict__ qkv,
                                                   float* __restrict__ out)
{
    const int bt = blockIdx.x;     // 0..23
    const int hh = blockIdx.y;     // 0..7
    __shared__ float Ks[64 * DHc];
    __shared__ float Vs[64 * DHc];
    const int t = threadIdx.x;
    const float scale = 0.125f;    // 1/sqrt(64)
    const float* base = qkv + (size_t)bt * Tc * (3 * Hc * DHc);

    float q[DHc], acc[DHc];
#pragma unroll
    for (int j = 0; j < DHc; ++j) {
        q[j] = base[(size_t)t * 1536 + hh * DHc + j];
        acc[j] = 0.f;
    }
    float mx = -1e30f, l = 0.f;

    for (int s0 = 0; s0 < Tc; s0 += 64) {
        __syncthreads();           // protect LDS from previous chunk's readers
#pragma unroll
        for (int j = 0; j < 4; ++j) {
            int chunk = (int)threadIdx.x + j * 256;   // 1024 x 16B per buffer
            int sr = chunk >> 4;
            int cc = (chunk & 15) << 2;
            async_load_b128(base + (size_t)(s0 + sr) * 1536 + 512  + hh * DHc + cc,
                            &Ks[sr * DHc + cc]);
            async_load_b128(base + (size_t)(s0 + sr) * 1536 + 1024 + hh * DHc + cc,
                            &Vs[sr * DHc + cc]);
        }
        wait_async();
        __syncthreads();
        for (int s = 0; s < 64; ++s) {
            float sc = 0.f;
#pragma unroll
            for (int j = 0; j < DHc; ++j) sc += q[j] * Ks[s * DHc + j];
            sc *= scale;
            float nm = fmaxf(mx, sc);
            float corr = __expf(mx - nm);
            float p = __expf(sc - nm);
            l = l * corr + p;
#pragma unroll
            for (int j = 0; j < DHc; ++j) acc[j] = acc[j] * corr + p * Vs[s * DHc + j];
            mx = nm;
        }
    }
    float inv = 1.f / l;
    float* o = out + ((size_t)bt * Tc + t) * Dc + hh * DHc;
#pragma unroll
    for (int j = 0; j < DHc; ++j) o[j] = acc[j] * inv;
}

// depthwise 3x3 'SAME' conv over the (F=3, T=256) grid, per channel
__global__ void head_conv_kernel(const float* __restrict__ x,
                                 const float* __restrict__ w,
                                 const float* __restrict__ cb,
                                 float* __restrict__ out)
{
    size_t idx = (size_t)blockIdx.x * blockDim.x + threadIdx.x;
    if (idx >= (size_t)BNc * Dc) return;
    int c = idx % Dc;
    size_t r = idx / Dc;
    int t = r % Tc;
    int f = (r / Tc) % Fc;
    int bb = r / (Tc * Fc);
    float s = cb[c];
#pragma unroll
    for (int i = 0; i < 3; ++i) {
        int ff = f + i - 1;
        if (ff < 0 || ff >= Fc) continue;
#pragma unroll
        for (int j = 0; j < 3; ++j) {
            int tt = t + j - 1;
            if (tt < 0 || tt >= Tc) continue;
            s += w[c * 9 + i * 3 + j] *
                 x[(((size_t)(bb * Fc + ff) * Tc + tt) * Dc) + c];
        }
    }
    out[idx] = s;
}

// ---------------------------------------------------------------------------
// host orchestration
// ---------------------------------------------------------------------------
static inline unsigned blk(size_t n) { return (unsigned)((n + 255) / 256); }

// convert A (f32, strided) -> bA (compact bf16), then bf16 WMMA GEMM
static inline void run_gemm(const float* A32, int lda,
                            const bf16* Bt,
                            const float* bias, int mode, int biasld,
                            const float* Cadd, float* C, int ldc,
                            int M, int N, int K, bf16* bA, hipStream_t s)
{
    cvt_bf16_kernel<<<blk((size_t)M * K), 256, 0, s>>>(A32, lda, bA, M, K);
    dim3 grid((unsigned)(((M / 16) * (N / 16) + 7) / 8));
    gemm_bf16_kernel<<<grid, 256, 0, s>>>(bA, Bt, bias, mode, biasld,
                                          Cadd, C, ldc, M, N, K);
}

extern "C" void kernel_launch(void* const* d_in, const int* in_sizes, int n_in,
                              void* d_out, int out_size, void* d_ws, size_t ws_size,
                              hipStream_t stream)
{
    (void)in_sizes; (void)n_in; (void)out_size; (void)ws_size;

    const int*   s_M      = (const int*)d_in[0];
    const float* tok_emb  = (const float*)d_in[1];
    const float* pos_emb  = (const float*)d_in[2];
    const float* out_bias = (const float*)d_in[3];
    const float* m_norm_w = (const float*)d_in[4];
    const float* m_Win    = (const float*)d_in[5];
    const float* m_convw  = (const float*)d_in[6];
    const float* m_convb  = (const float*)d_in[7];
    const float* m_Wx     = (const float*)d_in[8];
    const float* m_Wdt    = (const float*)d_in[9];
    const float* m_bdt    = (const float*)d_in[10];
    const float* m_Alog   = (const float*)d_in[11];
    const float* m_D      = (const float*)d_in[12];
    const float* m_Wout   = (const float*)d_in[13];
    const float* t_ln1_w  = (const float*)d_in[14];
    const float* t_ln1_b  = (const float*)d_in[15];
    const float* t_Wqkv   = (const float*)d_in[16];
    const float* t_Wo     = (const float*)d_in[17];
    const float* t_ln2_w  = (const float*)d_in[18];
    const float* t_ln2_b  = (const float*)d_in[19];
    const float* t_W1     = (const float*)d_in[20];
    const float* t_b1     = (const float*)d_in[21];
    const float* t_W2     = (const float*)d_in[22];
    const float* t_b2     = (const float*)d_in[23];
    const float* h_convw  = (const float*)d_in[24];
    const float* h_convb  = (const float*)d_in[25];
    const float* h_W      = (const float*)d_in[26];
    const float* h_b      = (const float*)d_in[27];
    const float* h_norm_w = (const float*)d_in[28];

    // ---- fp32 workspace arena ----
    const size_t XN  = (size_t)BNc * Dc;        // 3,145,728
    const size_t BIG = (size_t)BNc * 2 * DIc;   // 12,582,912
    const size_t UD  = (size_t)BNc * DIc;       // 6,291,456
    float* ws     = (float*)d_ws;
    float* w_x    = ws;                 // residual stream
    float* w_h    = w_x    + XN;        // normed activations
    float* w_skip = w_h    + XN;
    float* w_t2   = w_skip + XN;        // flipped h / per-dir Wout out / conv out
    float* w_big  = w_t2   + XN;        // uz / qkv / mlp-mid
    float* w_u    = w_big  + BIG;       // conv-silu u / attention out
    float* w_dt   = w_u    + UD;
    float* w_y    = w_dt   + UD;
    float* w_xdb  = w_y    + UD;        // BN x 64
    float* f32_end = w_xdb + (size_t)BNc * 64;

    // ---- bf16 arena: pre-transposed weights + activation staging ----
    bf16* bws   = (bf16*)f32_end;
    bf16* bWin  = bws;                                  // 4 x (2048 x 512)
    bf16* bWx   = bWin  + (size_t)4 * 2048 * 512;       // 4 x (64 x 1024)
    bf16* bWdt  = bWx   + (size_t)4 * 64 * 1024;        // 4 x (1024 x 32)
    bf16* bWout = bWdt  + (size_t)4 * 1024 * 32;        // 4 x (512 x 1024)
    bf16* bWqkv = bWout + (size_t)4 * 512 * 1024;       // 4 x (1536 x 512)
    bf16* bWo   = bWqkv + (size_t)4 * 1536 * 512;       // 4 x (512 x 512)
    bf16* bW1   = bWo   + (size_t)4 * 512 * 512;        // 4 x (2048 x 512)
    bf16* bW2   = bW1   + (size_t)4 * 2048 * 512;       // 4 x (512 x 2048)
    bf16* bhW   = bW2   + (size_t)4 * 512 * 2048;       // 512 x 512
    bf16* btok  = bhW   + (size_t)512 * 512;            // 1024 x 512 (already N x K)
    bf16* bA    = btok  + (size_t)1024 * 512;           // 6144 x 2048 staging

    // ---- one-time (per launch) weight conversion to transposed bf16 ----
    cvt_bf16_bt_kernel<<<blk((size_t)4*512*2048), 256, 0, stream>>>(m_Win,  bWin,  4, 512, 2048);
    cvt_bf16_bt_kernel<<<blk((size_t)4*1024*64),  256, 0, stream>>>(m_Wx,   bWx,   4, 1024, 64);
    cvt_bf16_bt_kernel<<<blk((size_t)4*32*1024),  256, 0, stream>>>(m_Wdt,  bWdt,  4, 32, 1024);
    cvt_bf16_bt_kernel<<<blk((size_t)4*1024*512), 256, 0, stream>>>(m_Wout, bWout, 4, 1024, 512);
    cvt_bf16_bt_kernel<<<blk((size_t)4*512*1536), 256, 0, stream>>>(t_Wqkv, bWqkv, 4, 512, 1536);
    cvt_bf16_bt_kernel<<<blk((size_t)4*512*512),  256, 0, stream>>>(t_Wo,   bWo,   4, 512, 512);
    cvt_bf16_bt_kernel<<<blk((size_t)4*512*2048), 256, 0, stream>>>(t_W1,   bW1,   4, 512, 2048);
    cvt_bf16_bt_kernel<<<blk((size_t)4*2048*512), 256, 0, stream>>>(t_W2,   bW2,   4, 2048, 512);
    cvt_bf16_bt_kernel<<<blk((size_t)512*512),    256, 0, stream>>>(h_W,    bhW,   1, 512, 512);
    cvt_bf16_kernel   <<<blk((size_t)1024*512),   256, 0, stream>>>(tok_emb, 512, btok, 1024, 512);

    // ---- embedding ----
    embed_kernel<<<blk(XN), 256, 0, stream>>>(s_M, tok_emb, pos_emb, w_x);

    // ---- bidirectional Mamba layers ----
    for (int l = 0; l < 2; ++l) {
        rmsnorm_kernel<<<BNc, 256, 0, stream>>>(w_x, m_norm_w + l * Dc, w_h, 1e-5f);
        for (int dir = 0; dir < 2; ++dir) {
            const int ld = l * 2 + dir;
            const float* hin = w_h;
            if (dir == 1) {
                flip_kernel<<<blk(XN), 256, 0, stream>>>(w_h, w_t2, Dc);
                hin = w_t2;
            }
            // uz = h @ Win  (512 -> 2048)
            run_gemm(hin, Dc, bWin + (size_t)ld * 2048 * 512,
                     nullptr, EPI_STORE, 0, nullptr, w_big, 2 * DIc,
                     BNc, 2 * DIc, Dc, bA, stream);
            // causal depthwise conv + silu
            mamba_conv_kernel<<<blk(UD), 256, 0, stream>>>(
                w_big, m_convw + (size_t)ld * DIc * Kc, m_convb + (size_t)ld * DIc, w_u);
            // xdb = u @ Wx  (1024 -> 64)
            run_gemm(w_u, DIc, bWx + (size_t)ld * 64 * 1024,
                     nullptr, EPI_STORE, 0, nullptr, w_xdb, 64,
                     BNc, 64, DIc, bA, stream);
            // dt = softplus(xdb[:, :32] @ Wdt + bdt)  (32 -> 1024)
            run_gemm(w_xdb, 64, bWdt + (size_t)ld * 1024 * 32,
                     m_bdt + (size_t)ld * DIc, EPI_BIAS_SOFTPLUS, 0,
                     nullptr, w_dt, DIc, BNc, DIc, DTRc, bA, stream);
            // recurrent scan
            mamba_scan_kernel<<<blk(Bc * DIc), 256, 0, stream>>>(
                w_dt, w_u, w_xdb, m_Alog + (size_t)ld * DIc * DSc, w_y);
            // y = (y + D*u) * silu(z)
            gate_kernel<<<blk(UD), 256, 0, stream>>>(w_y, w_u, w_big,
                                                     m_D + (size_t)ld * DIc);
            // out = y @ Wout  (1024 -> 512), residual add (flipped for dir 1)
            if (dir == 0) {
                run_gemm(w_y, DIc, bWout + (size_t)ld * 512 * 1024,
                         nullptr, EPI_ADD, 0, w_x, w_x, Dc,
                         BNc, Dc, DIc, bA, stream);
            } else {
                run_gemm(w_y, DIc, bWout + (size_t)ld * 512 * 1024,
                         nullptr, EPI_STORE, 0, nullptr, w_t2, Dc,
                         BNc, Dc, DIc, bA, stream);
                addflip_kernel<<<blk(XN), 256, 0, stream>>>(w_t2, w_x);
            }
        }
    }

    // ---- save skip ----
    hipMemcpyAsync(w_skip, w_x, XN * sizeof(float),
                   hipMemcpyDeviceToDevice, stream);

    // ---- transformer stack (x viewed as (24, 256, 512): same memory) ----
    for (int d = 0; d < DEPTHc; ++d) {
        layernorm_kernel<<<BNc, 256, 0, stream>>>(w_x, t_ln1_w + d * Dc,
                                                  t_ln1_b + d * Dc, w_h, 1e-5f);
        run_gemm(w_h, Dc, bWqkv + (size_t)d * 1536 * 512,
                 nullptr, EPI_STORE, 0, nullptr, w_big, 1536,
                 BNc, 1536, Dc, bA, stream);
        attn_kernel<<<dim3(Bc * Fc, Hc), 256, 0, stream>>>(w_big, w_u);
        run_gemm(w_u, Dc, bWo + (size_t)d * 512 * 512,
                 nullptr, EPI_ADD, 0, w_x, w_x, Dc, BNc, Dc, Dc, bA, stream);
        layernorm_kernel<<<BNc, 256, 0, stream>>>(w_x, t_ln2_w + d * Dc,
                                                  t_ln2_b + d * Dc, w_h, 1e-5f);
        run_gemm(w_h, Dc, bW1 + (size_t)d * 2048 * 512,
                 t_b1 + (size_t)d * 4 * Dc, EPI_BIAS_GELU, 0,
                 nullptr, w_big, 4 * Dc, BNc, 4 * Dc, Dc, bA, stream);
        run_gemm(w_big, 4 * Dc, bW2 + (size_t)d * 512 * 2048,
                 t_b2 + (size_t)d * Dc, EPI_BIAS_ADD, 0,
                 w_x, w_x, Dc, BNc, Dc, 4 * Dc, bA, stream);
    }

    // ---- head ----
    add_kernel<<<blk(XN), 256, 0, stream>>>(w_x, w_skip);
    head_conv_kernel<<<blk(XN), 256, 0, stream>>>(w_x, h_convw, h_convb, w_t2);
    run_gemm(w_t2, Dc, bhW, h_b, EPI_BIAS_GELU, 0,
             nullptr, w_h, Dc, BNc, Dc, Dc, bA, stream);
    rmsnorm_kernel<<<BNc, 256, 0, stream>>>(w_h, h_norm_w, w_t2, 1e-4f);
    // logits = x @ tok_emb^T + out_bias  (only first 1024 vocab columns kept)
    run_gemm(w_t2, Dc, btok, out_bias, EPI_OUTBIAS, CBc + 1,
             nullptr, (float*)d_out, CBc, BNc, CBc, Dc, bA, stream);
}
